// SegformerModelLossSemsegGatedCRF_6468220748156
// MI455X (gfx1250) — matrix operand order repaired and checked by reading, not compiled, not run on hardware
//
#include <hip/hip_runtime.h>

// CDNA5 WMMA fragment types (wave32)
typedef __attribute__((ext_vector_type(16))) _Float16 v16h;
typedef __attribute__((ext_vector_type(8)))  float    v8f;
// TDM descriptor group types
typedef __attribute__((ext_vector_type(4))) unsigned int v4u;
typedef __attribute__((ext_vector_type(4))) int          v4i;
typedef __attribute__((ext_vector_type(8))) int          v8i;

#if __has_builtin(__builtin_amdgcn_tensor_load_to_lds) && \
    __has_builtin(__builtin_amdgcn_s_wait_tensorcnt)
#define USE_TDM 1
#else
#define USE_TDM 0
#endif

// Problem constants from the reference (static under jit)
constexpr int RAD  = 5;
constexpr int DIA  = 11;        // 2*RAD+1
constexpr int NB   = 4;         // N
constexpr int CC   = 21;        // C (softmax channels)
constexpr int HH   = 96;
constexpr int WW   = 96;
constexpr int CPAD = 32;        // channels padded to WMMA K=32
constexpr int COLS = 32;        // staged columns: x0-5 .. x0+26
constexpr int NDOT = 16 * DIA * DIA;  // banded dot entries per tile

// f32 TDM staging (ch-major, TDM packs tile contiguously: [z=ch][y=row][x=col])
// overlapped with the feature / dot arrays, which are only live afterwards.
union SharedPool {
    float stage[CC * DIA * COLS];                      // 29568 B
    struct {
        float fS[DIA][COLS][6];                        //  8448 B
        float dotPad[NDOT + 8];                        //  7776 B (+dump slot)
    } p;
};

__global__ __launch_bounds__(32)
void gatedcrf_wmma_kernel(const float* __restrict__ ysm,
                          const float* __restrict__ fxy,
                          const float* __restrict__ frgb,
                          float* __restrict__ out)
{
    __shared__ _Float16   yS[DIA][COLS][CPAD];         // 22528 B, f16 WMMA operand tile
    __shared__ SharedPool sp;                          // 29568 B  -> total ~51 KB

    const int lane = threadIdx.x;                      // wave32: one wave per block
    const int x0   = blockIdx.x * 16;                  // center tile base column
    const int yr   = blockIdx.y;                       // image row
    const int n    = blockIdx.z;                       // batch

    // ---------------- Phase A0: stage y tile into yS (f16, zero padded) -----
    const bool interior = (x0 >= 16) && (x0 + 26 < WW) &&
                          (yr >= RAD) && (yr + RAD < HH);
#if USE_TDM
    if (interior) {
        // Tensor Data Mover: 3D tile (x=32, y=11, z=21) from NCHW f32 tensor.
        // Block-uniform descriptor -> SGPR groups.
        const unsigned long long gaddr =
            (unsigned long long)(ysm + (((size_t)n * CC) * HH + (yr - RAD)) * WW
                                     + (x0 - RAD));
        const unsigned int ldsa = (unsigned int)(size_t)(void*)sp.stage;

        v4u g0;
        g0[0] = 1u;                                   // count=1 (valid D#)
        g0[1] = ldsa;                                 // lds_addr [63:32]
        g0[2] = (unsigned int)gaddr;                  // global_addr lo
        g0[3] = (unsigned int)((gaddr >> 32) & 0x01FFFFFFu) | (2u << 30); // type=2

        v8i g1;
        g1[0] = (2 << 16);                            // data_size=2 -> 4 bytes
        g1[1] = (WW << 16);                           // tensor_dim0 lo16 @ [63:48]
        g1[2] = (HH << 16);                           // tensor_dim1 lo16 @ [95:80]
        g1[3] = (COLS << 16);                         // tile_dim0 @ [127:112]
        g1[4] = DIA | (CC << 16);                     // tile_dim1 | tile_dim2
        g1[5] = WW;                                   // tensor_dim0_stride lo32
        g1[6] = (int)((unsigned)(HH * WW) << 16);     // dim1_stride lo16 @ [223:208]
        g1[7] = (int)((unsigned)(HH * WW) >> 16);     // dim1_stride bits [255:224]

        v4i g2; g2[0] = CC; g2[1] = 0; g2[2] = 0; g2[3] = 0;  // tensor_dim2=21
        v4i g3; g3[0] = 0;  g3[1] = 0; g3[2] = 0; g3[3] = 0;

#if defined(__clang_major__) && (__clang_major__ >= 23)
        v8i g4 = {};
        __builtin_amdgcn_tensor_load_to_lds(g0, g1, g2, g3, g4, 0);
#else
        __builtin_amdgcn_tensor_load_to_lds(g0, g1, g2, g3, 0);
#endif
        __builtin_amdgcn_s_wait_tensorcnt(0);

        // repack f32 [ch][ry][col] -> f16 [ry][col][ch], zero channels >= 21
        for (int t = lane; t < DIA * COLS * CPAD; t += 32) {
            const int ch  = t & (CPAD - 1);
            const int col = (t >> 5) & (COLS - 1);
            const int ry  = t >> 10;
            const float v = (ch < CC) ? sp.stage[(ch * DIA + ry) * COLS + col]
                                      : 0.0f;
            yS[ry][col][ch] = (_Float16)v;
        }
    } else
#endif
    {
        // scalar fallback (boundary tiles): zero pad OOB pixels / channels
        for (int rc = 0; rc < DIA * CPAD; ++rc) {
            const int ry = rc / CPAD;
            const int ch = rc % CPAD;
            const int gy = yr + ry - RAD;
            const int gx = x0 - RAD + lane;
            float v = 0.0f;
            if (ch < CC && gy >= 0 && gy < HH && gx >= 0 && gx < WW)
                v = ysm[(((size_t)n * CC + ch) * HH + gy) * WW + gx];
            yS[ry][lane][ch] = (_Float16)v;
        }
    }
    __syncthreads();   // stage buffer dead beyond this point; union reuse is safe

    // ---------------- Phase A1: normalized features + dot band clear --------
    for (int ry = 0; ry < DIA; ++ry) {
        const int gy = yr + ry - RAD;
        const int gx = x0 - RAD + lane;
        float vx = 0.f, vy = 0.f, rr = 0.f, gg = 0.f, bb = 0.f;
        if (gy >= 0 && gy < HH && gx >= 0 && gx < WW) {
            const size_t b2 = (((size_t)n * 2) * HH + gy) * WW + gx;
            vx = fxy[b2];
            vy = fxy[b2 + (size_t)HH * WW];
            const size_t b3 = (((size_t)n * 3) * HH + gy) * WW + gx;
            rr = frgb[b3];
            gg = frgb[b3 + (size_t)HH * WW];
            bb = frgb[b3 + 2 * (size_t)HH * WW];
            const float n2 = sqrtf(vx * vx + vy * vy);
            const float s2 = 1.0f / (fmaxf(n2, 1e-12f) * 6.0f);
            vx *= s2; vy *= s2;
            const float n3 = sqrtf(rr * rr + gg * gg + bb * bb);
            const float s3 = 1.0f / (fmaxf(n3, 1e-12f) * 6.0f);
            rr *= s3; gg *= s3; bb *= s3;
        }
        sp.p.fS[ry][lane][0] = vx; sp.p.fS[ry][lane][1] = vy;
        sp.p.fS[ry][lane][2] = rr; sp.p.fS[ry][lane][3] = gg;
        sp.p.fS[ry][lane][4] = bb; sp.p.fS[ry][lane][5] = 0.0f;
    }
    for (int t = lane; t < NDOT + 8; t += 32)
        sp.p.dotPad[t] = 0.0f;
    __syncthreads();

    // ---------------- Phase B: banded Gram via V_WMMA_F32_16X16X32_F16 ------
    const int kg = lane >> 4;     // K-group: 0 for lanes 0-15, 1 for 16-31
    const int mN = lane & 15;     // M (for A) / N (for B/D) index

    // A fragment 16x32 (MxK): lane<16: VGPR0-3 -> K 0..7, VGPR4-7 -> K 16..23;
    //                         lane>=16: VGPR0-3 -> K 8..15, VGPR4-7 -> K 24..31.
    v16h afrag;
    {
        const int colA = mN + RAD;
        for (int i = 0; i < 8; ++i) {
            const int k = (i < 4) ? (kg * 8 + 2 * i)
                                  : (16 + kg * 8 + 2 * (i - 4));
            afrag[2 * i]     = yS[RAD][colA][k];
            afrag[2 * i + 1] = yS[RAD][colA][k + 1];
        }
    }

    for (int ry = 0; ry < DIA; ++ry) {
        const int gy = yr + ry - RAD;
        if (gy < 0 || gy >= HH) continue;      // uniform; dot entries stay 0
        for (int bt = 0; bt < 2; ++bt) {
            // B fragment 32x16 (KxN): same K order as A (lanes 0-15: K 0..7,16..23; lanes 16-31: K 8..15,24..31)
            v16h bfrag;
            const int cs = bt * 16 + mN;
            for (int t = 0; t < 8; ++t) {
                const int k0 = (t < 4) ? (kg * 8 + 2 * t) : (16 + kg * 8 + 2 * (t - 4));
                bfrag[2 * t]     = yS[ry][cs][k0];
                bfrag[2 * t + 1] = yS[ry][cs][k0 + 1];
            }
            v8f dacc = {};
            dacc = __builtin_amdgcn_wmma_f32_16x16x32_f16(
                false, afrag, false, bfrag, (short)0, dacc, false, false);
            // Branchless banded scatter: out-of-band elements go to a dump slot
            // (select -> v_cndmask, single unconditional ds_store per element).
            for (int i = 0; i < 8; ++i) {
                const int m   = i + kg * 8;
                const unsigned dxi =
                    (unsigned)(bt * 16 - RAD + mN - m + RAD);   // dx + RAD
                const int off = (dxi <= 10u)
                              ? (m * (DIA * DIA) + ry * DIA + (int)dxi)
                              : NDOT;                            // dump
                sp.p.dotPad[off] = dacc[i];
            }
        }
    }
    __syncthreads();

    // ---------------- Phase C: sum K(p,j)*(1 - dot) --------------------------
    float acc = 0.0f;
    for (int t = lane; t < NDOT; t += 32) {
        const int m   = t / (DIA * DIA);
        const int j   = t % (DIA * DIA);
        const int ry  = j / DIA;
        const int dxi = j % DIA;
        if (ry == RAD && dxi == RAD) continue;     // center kernel entry = 0
        const float* fp = sp.p.fS[RAD][m + RAD];   // center features
        const float* fq = sp.p.fS[ry][m + dxi];    // neighbor (OOB staged as 0)
        const float d0 = fp[0] - fq[0];
        const float d1 = fp[1] - fq[1];
        const float exy = __expf(-0.5f * (d0 * d0 + d1 * d1));
        const float d2 = fp[2] - fq[2];
        const float d3 = fp[3] - fq[3];
        const float d4 = fp[4] - fq[4];
        const float ergb = __expf(-0.5f * (d2 * d2 + d3 * d3 + d4 * d4));
        const float K = 0.1f * exy + 0.2f * ergb;
        acc += K * (1.0f - sp.p.dotPad[t]);
    }
    for (int off = 16; off > 0; off >>= 1)
        acc += __shfl_down(acc, off, 32);
    if (lane == 0)
        atomicAdd(out, acc * (1.0f / (float)(NB * HH * WW)));
}

extern "C" void kernel_launch(void* const* d_in, const int* in_sizes, int n_in,
                              void* d_out, int out_size, void* d_ws, size_t ws_size,
                              hipStream_t stream)
{
    const float* ysm  = (const float*)d_in[0];   // y_hat_softmax (4,21,96,96) f32
    const float* fxy  = (const float*)d_in[1];   // feat_xy       (4, 2,96,96) f32
    const float* frgb = (const float*)d_in[2];   // feat_rgb      (4, 3,96,96) f32
    // d_in[3] = kernels_radius (==5, baked in as RAD)
    float* out = (float*)d_out;                  // scalar loss

    hipMemsetAsync(out, 0, sizeof(float), stream);
    dim3 grid(WW / 16, HH, NB);                  // (6, 96, 4) tiles
    gatedcrf_wmma_kernel<<<grid, 32, 0, stream>>>(ysm, fxy, frgb, out);
}
